// MambaCriticNetwork_58059367907938
// MI455X (gfx1250) — matrix operand order repaired and checked by reading, not compiled, not run on hardware
//
#include <hip/hip_runtime.h>
#include <hip/hip_bf16.h>
#include <math.h>

// ---------------- problem constants ----------------
#define B_    32
#define S_    2048
#define F_    64
#define H_    256
#define DI_   512
#define KCONV 4
#define NL_   4
#define CH_   8                      // scan chunks
#define LCH_  (S_ / CH_)             // 256 steps per chunk
#define MBS   ((size_t)B_ * S_)      // 65536 rows

typedef _Float16 h16_t;
typedef h16_t v16h __attribute__((ext_vector_type(16)));
typedef h16_t v8h  __attribute__((ext_vector_type(8)));
typedef float v8f  __attribute__((ext_vector_type(8)));

__device__ __forceinline__ float silu_f(float x) { return x / (1.f + __expf(-x)); }

// -------- WMMA fragment loaders (CDNA5 16x16x32 f16 layout) --------
__device__ __forceinline__ v16h frag_a(const h16_t* p /*row start + kb*/, int half) {
  v8h lo = *(const v8h*)(p + half * 8);
  v8h hi = *(const v8h*)(p + 16 + half * 8);
  v16h f;
#pragma unroll
  for (int i = 0; i < 8; ++i) { f[i] = lo[i]; f[i + 8] = hi[i]; }
  return f;
}
__device__ __forceinline__ v16h frag_b(const h16_t* p /*col start + kb*/, int half) {
  v8h lo = *(const v8h*)(p + half * 8);
  v8h hi = *(const v8h*)(p + 16 + half * 8);
  v16h f;
#pragma unroll
  for (int i = 0; i < 8; ++i) { f[i] = lo[i]; f[i + 8] = hi[i]; }
  return f;
}

// -------- async global->LDS tile copy (CDNA5 ASYNCcnt path) --------
// Copies `bytes` contiguous bytes (multiple of 16) from gsrc into lds.
// Uses GLOBAL_LOAD_ASYNC_TO_LDS_B128; completion via s_wait_asynccnt.
__device__ __forceinline__ void async_tile_copy(const h16_t* __restrict__ gsrc,
                                                h16_t* __restrict__ lds,
                                                int bytes)
{
  const uint32_t lbase = (uint32_t)(uintptr_t)lds;   // low 32 bits = LDS offset
  const uint64_t gbase = (uint64_t)(uintptr_t)gsrc;
  for (int i = threadIdx.x * 16; i < bytes; i += blockDim.x * 16) {
    uint32_t la = lbase + (uint32_t)i;
    uint64_t ga = gbase + (uint64_t)i;
    asm volatile("global_load_async_to_lds_b128 %0, %1, off"
                 :: "v"(la), "v"(ga) : "memory");
  }
  asm volatile("s_wait_asynccnt 0x0" ::: "memory");
  __syncthreads();
}

// ---------------------------------------------------------------
// GEMM with elementwise epilogue. 16 waves; block tile 16M x 256N.
// A tile (16 x K, contiguous) staged in LDS via async copy.
// mode 0: in_proj  -> silu(xi) f32, silu(res) f16
// mode 1: dt proj  -> sigmoid(softplus(z)) f16 (alpha)
// ---------------------------------------------------------------
__global__ void __launch_bounds__(512)
gemm_epi_kernel(const h16_t* __restrict__ A, const h16_t* __restrict__ Bt,
                const float* __restrict__ bias, int K, int mode,
                float* __restrict__ outXI, h16_t* __restrict__ outSRES,
                h16_t* __restrict__ outALPHA)
{
  __shared__ h16_t tA[16 * 512];
  const int lane = threadIdx.x & 31;
  const int wave = threadIdx.x >> 5;     // 0..15
  const int half = lane >> 4;
  const int l15  = lane & 15;
  const size_t rowBase = (size_t)blockIdx.x * 16;
  const int    colBase = blockIdx.y * 256 + wave * 16;

  async_tile_copy(A + rowBase * (size_t)K, tA, 16 * K * 2);

  const h16_t* arow = tA + l15 * K;
  const h16_t* brow = Bt + (size_t)(colBase + l15) * K;

  v8f acc = {};
  for (int kb = 0; kb < K; kb += 32) {
    v16h a = frag_a(arow + kb, half);
    v16h b = frag_b(brow + kb, half);
    acc = __builtin_amdgcn_wmma_f32_16x16x32_f16(false, a, false, b,
                                                 (short)0, acc, false, false);
  }

  const int   cg = colBase + l15;
  const float bv = bias[cg];
#pragma unroll
  for (int i = 0; i < 8; ++i) {
    const size_t row = rowBase + i + 8 * half;
    float z = acc[i] + bv;
    if (mode == 0) {
      float v = silu_f(z);
      if (cg < DI_) outXI[row * DI_ + cg] = v;                  // silu(xi)
      else          outSRES[row * DI_ + (cg - DI_)] = (h16_t)v; // silu(res)
    } else {
      float sp = (z > 20.f) ? z : log1pf(__expf(z));            // softplus
      float al = 1.f / (1.f + __expf(-sp));                     // sigmoid
      outALPHA[row * DI_ + cg] = (h16_t)al;
    }
  }
}

// ---------------------------------------------------------------
// GEMM (full 256-wide row per block, 16 waves) fused with
// bias (+optional residual) + LayerNorm (+optional ReLU).
// A tile staged in LDS via async copy.
// ---------------------------------------------------------------
__global__ void __launch_bounds__(512)
gemm_ln_kernel(const h16_t* __restrict__ A, const h16_t* __restrict__ Bt, int K,
               const float* __restrict__ bias, const float* __restrict__ gamma,
               const float* __restrict__ beta, const float* __restrict__ resid,
               int relu, float* __restrict__ outX, h16_t* __restrict__ outX16)
{
  __shared__ h16_t tA[16 * 512];
  __shared__ float tile[16][264];
  const int lane = threadIdx.x & 31;
  const int wave = threadIdx.x >> 5;    // 0..15, owns 16 output columns
  const int half = lane >> 4;
  const int l15  = lane & 15;
  const size_t rowBase = (size_t)blockIdx.x * 16;
  const int    colBase = wave * 16;

  async_tile_copy(A + rowBase * (size_t)K, tA, 16 * K * 2);

  const h16_t* arow = tA + l15 * K;
  const h16_t* brow = Bt + (size_t)(colBase + l15) * K;

  v8f acc = {};
  for (int kb = 0; kb < K; kb += 32) {
    v16h a = frag_a(arow + kb, half);
    v16h b = frag_b(brow + kb, half);
    acc = __builtin_amdgcn_wmma_f32_16x16x32_f16(false, a, false, b,
                                                 (short)0, acc, false, false);
  }

  const int   cg = colBase + l15;
  const float bv = bias[cg];
#pragma unroll
  for (int i = 0; i < 8; ++i) {
    const int r = i + 8 * half;
    float z = acc[i] + bv;
    if (resid) z += resid[(rowBase + r) * H_ + cg];
    tile[r][cg] = z;
  }
  __syncthreads();

  // LayerNorm: wave w normalizes row w (256 values, 8 per lane)
  float s1 = 0.f, s2 = 0.f;
#pragma unroll
  for (int j = 0; j < 8; ++j) {
    float v = tile[wave][lane + 32 * j];
    s1 += v; s2 += v * v;
  }
#pragma unroll
  for (int off = 16; off; off >>= 1) {
    s1 += __shfl_xor(s1, off, 32);
    s2 += __shfl_xor(s2, off, 32);
  }
  const float mean = s1 * (1.f / H_);
  const float var  = s2 * (1.f / H_) - mean * mean;
  const float rstd = rsqrtf(var + 1e-5f);
  const size_t row = rowBase + wave;
#pragma unroll
  for (int j = 0; j < 8; ++j) {
    int   col = lane + 32 * j;
    float v = (tile[wave][col] - mean) * rstd * gamma[col] + beta[col];
    if (relu) v = fmaxf(v, 0.f);
    outX[row * H_ + col]   = v;
    outX16[row * H_ + col] = (h16_t)v;
  }
}

// ---- depthwise causal conv (K=4) + bias + silu -> f16 ----
__global__ void conv_silu_kernel(const float* __restrict__ XI,
                                 const float* __restrict__ cw,
                                 const float* __restrict__ cb,
                                 h16_t* __restrict__ out)
{
  size_t idx = (size_t)blockIdx.x * blockDim.x + threadIdx.x;
  if (idx >= MBS * DI_) return;
  int    c  = (int)(idx % DI_);
  size_t bs = idx / DI_;
  int    s  = (int)(bs % S_);
  float acc = cb[c];
#pragma unroll
  for (int j = 0; j < KCONV; ++j) {
    int ss = s - (KCONV - 1) + j;
    if (ss >= 0) acc += cw[c * KCONV + j] * XI[(bs + (size_t)(ss - s)) * DI_ + c];
  }
  out[idx] = (h16_t)silu_f(acc);
}

// ---- chunk-parallel linear scan: y_t = a'_t x_t + (1-a'_t) y_{t-1},
//      a'_0 := 1. Chunked into CH_ chunks of LCH_ steps.            ----
// pass A: per (b, chunk, c) compute P = prod(1-a'), Q = local scan
__global__ void scanA_kernel(const h16_t* __restrict__ X2,
                             const h16_t* __restrict__ AL,
                             float* __restrict__ Pb, float* __restrict__ Qb)
{
  int t = blockIdx.x * blockDim.x + threadIdx.x;
  if (t >= B_ * CH_ * DI_) return;
  int c = t % DI_;
  int g = (t / DI_) % CH_;
  int b = t / (DI_ * CH_);
  size_t id = ((size_t)b * S_ + (size_t)g * LCH_) * DI_ + c;
  float P = 1.f, Q = 0.f;
  for (int s = 0; s < LCH_; ++s, id += DI_) {
    int   sg = g * LCH_ + s;
    float x  = (float)X2[id];
    float a  = (sg == 0) ? 1.f : (float)AL[id];
    Q = a * x + (1.f - a) * Q;
    P *= (1.f - a);
  }
  Pb[t] = P; Qb[t] = Q;
}
// pass B: per (b, c) combine the CH_ chunk summaries -> y_in per chunk
__global__ void scanB_kernel(const float* __restrict__ Pb,
                             const float* __restrict__ Qb,
                             float* __restrict__ Yin)
{
  int t = blockIdx.x * blockDim.x + threadIdx.x;
  if (t >= B_ * DI_) return;
  int c = t % DI_, b = t / DI_;
  float y = 0.f;
  for (int g = 0; g < CH_; ++g) {
    int idx = (b * CH_ + g) * DI_ + c;
    Yin[idx] = y;
    y = Pb[idx] * y + Qb[idx];
  }
}
// pass C: re-scan each chunk with correct y_in; fuse (+x*D) * silu(res)
__global__ void scanC_kernel(const h16_t* __restrict__ X2,
                             const h16_t* __restrict__ AL,
                             const h16_t* __restrict__ SR,
                             const float* __restrict__ Dp,
                             const float* __restrict__ Yin,
                             h16_t* __restrict__ Y)
{
  int t = blockIdx.x * blockDim.x + threadIdx.x;
  if (t >= B_ * CH_ * DI_) return;
  int c = t % DI_;
  int g = (t / DI_) % CH_;
  int b = t / (DI_ * CH_);
  const float d = Dp[c];
  float y = Yin[t];
  size_t id = ((size_t)b * S_ + (size_t)g * LCH_) * DI_ + c;
  for (int s = 0; s < LCH_; ++s, id += DI_) {
    int   sg = g * LCH_ + s;
    float x  = (float)X2[id];
    float a  = (sg == 0) ? 1.f : (float)AL[id];
    y = a * x + (1.f - a) * y;
    float o = (y + x * d) * (float)SR[id];
    Y[id] = (h16_t)o;
  }
}

// ---- small helpers ----
__global__ void f32_to_f16_kernel(const float* __restrict__ src,
                                  h16_t* __restrict__ dst, size_t n)
{
  size_t i = (size_t)blockIdx.x * blockDim.x + threadIdx.x;
  if (i < n) dst[i] = (h16_t)src[i];
}
// src: K x N (row major f32)  ->  dst: N x K (f16), i.e. B^T for WMMA
__global__ void wt_transpose_kernel(const float* __restrict__ src,
                                    h16_t* __restrict__ dst, int K, int N)
{
  int i = blockIdx.x * blockDim.x + threadIdx.x;
  if (i >= K * N) return;
  int n = i / K, k = i % K;
  dst[i] = (h16_t)src[(size_t)k * N + n];
}

// ---- tiny 32-row MLP head (single block) ----
__device__ __forceinline__ float blockSum256(float v) {
  __shared__ float sc[8];
  __shared__ float res;
  int lane = threadIdx.x & 31, w = threadIdx.x >> 5;
#pragma unroll
  for (int off = 16; off; off >>= 1) v += __shfl_xor(v, off, 32);
  if (lane == 0) sc[w] = v;
  __syncthreads();
  if (w == 0) {
    float r = (lane < 8) ? sc[lane] : 0.f;
#pragma unroll
    for (int off = 16; off; off >>= 1) r += __shfl_xor(r, off, 32);
    if (lane == 0) res = r;
  }
  __syncthreads();
  return res;
}

__global__ void __launch_bounds__(256)
head_mlp_kernel(const float* __restrict__ X,
                const float* __restrict__ ps,  const float* __restrict__ act,
                const float* __restrict__ pfw, const float* __restrict__ pfb,
                const float* __restrict__ acw, const float* __restrict__ acb,
                const float* __restrict__ f1w, const float* __restrict__ f1b,
                const float* __restrict__ g1,  const float* __restrict__ b1,
                const float* __restrict__ f2w, const float* __restrict__ f2b,
                const float* __restrict__ g2,  const float* __restrict__ b2,
                const float* __restrict__ f3w, const float* __restrict__ f3b,
                float* __restrict__ out)
{
  __shared__ float h[384];
  __shared__ float h2[256];
  const int t = threadIdx.x;
  for (int b = 0; b < B_; ++b) {
    h[t] = X[((size_t)b * S_ + (S_ - 1)) * H_ + t];            // feat
    if (t < 64) {
      h[256 + t] = fmaxf(ps[b * 2 + 0] * pfw[t] + ps[b * 2 + 1] * pfw[64 + t] + pfb[t], 0.f);
      h[320 + t] = fmaxf(act[b] * acw[t] + acb[t], 0.f);
    }
    __syncthreads();

    float z = f1b[t];
    for (int k = 0; k < 384; ++k) z += h[k] * f1w[k * 256 + t];
    float s1 = blockSum256(z), s2 = blockSum256(z * z);
    float mean = s1 / 256.f, var = s2 / 256.f - mean * mean;
    float v = fmaxf((z - mean) * rsqrtf(var + 1e-5f) * g1[t] + b1[t], 0.f);
    h2[t] = v;
    __syncthreads();

    z = f2b[t];
    for (int k = 0; k < 256; ++k) z += h2[k] * f2w[k * 256 + t];
    s1 = blockSum256(z); s2 = blockSum256(z * z);
    mean = s1 / 256.f; var = s2 / 256.f - mean * mean;
    v = fmaxf((z - mean) * rsqrtf(var + 1e-5f) * g2[t] + b2[t], 0.f);

    float total = blockSum256(v * f3w[t]);
    if (t == 0) out[b] = total + f3b[0];
    __syncthreads();
  }
}

// ---------------------------------------------------------------
// Host orchestration
// ---------------------------------------------------------------
extern "C" void kernel_launch(void* const* d_in, const int* in_sizes, int n_in,
                              void* d_out, int out_size, void* d_ws, size_t ws_size,
                              hipStream_t stream)
{
  (void)in_sizes; (void)n_in; (void)out_size; (void)ws_size;
  const float* market = (const float*)d_in[0];
  const float* ps     = (const float*)d_in[1];
  const float* act    = (const float*)d_in[2];
  // params flattened depth-first: per head 62 leaves (emb 4, 4 blocks x 11, head 14)
  auto P = [&](int head, int idx) -> const float* {
    return (const float*)d_in[3 + head * 62 + idx];
  };

  char* wp = (char*)d_ws;
  auto alloc = [&](size_t bytes) -> char* {
    char* p = wp; wp += (bytes + 255) & ~(size_t)255; return p;
  };
  float* X     = (float*)alloc(MBS * H_  * 4);   // residual stream (f32)
  h16_t* X16   = (h16_t*)alloc(MBS * H_  * 2);   // residual stream (f16, GEMM A)
  float* XI    = (float*)alloc(MBS * DI_ * 4);   // silu(xi) pre-conv (f32)
  h16_t* ALPHA = (h16_t*)XI;                     // alias: alpha overwrites dead XI
  h16_t* SRES  = (h16_t*)alloc(MBS * DI_ * 2);   // silu(res)
  h16_t* XI2   = (h16_t*)alloc(MBS * DI_ * 2);   // silu(conv)
  h16_t* Y16   = (h16_t*)alloc(MBS * DI_ * 2);   // scan output (GEMM A)
  h16_t* MK16  = (h16_t*)alloc(MBS * F_  * 2);   // market f16
  float* Pb    = (float*)alloc((size_t)B_ * CH_ * DI_ * 4);
  float* Qb    = (float*)alloc((size_t)B_ * CH_ * DI_ * 4);
  float* Yin   = (float*)alloc((size_t)B_ * CH_ * DI_ * 4);
  h16_t* embWt = (h16_t*)alloc((size_t)H_ * F_  * 2);
  h16_t* inWt  = (h16_t*)alloc((size_t)2 * DI_ * H_ * 2);
  h16_t* dtWt  = (h16_t*)alloc((size_t)DI_ * DI_ * 2);
  h16_t* outWt = (h16_t*)alloc((size_t)H_ * DI_ * 2);

  { // market -> f16 (shared by both heads)
    size_t n = MBS * F_;
    f32_to_f16_kernel<<<(unsigned)((n + 255) / 256), 256, 0, stream>>>(market, MK16, n);
  }

  for (int head = 0; head < 2; ++head) {
    // ---- embed + LN + ReLU ----
    wt_transpose_kernel<<<(F_ * H_ + 255) / 256, 256, 0, stream>>>(P(head, 0), embWt, F_, H_);
    gemm_ln_kernel<<<(unsigned)(MBS / 16), 512, 0, stream>>>(
        MK16, embWt, F_, P(head, 1), P(head, 2), P(head, 3), nullptr, 1, X, X16);

    for (int l = 0; l < NL_; ++l) {
      const int bb = 4 + l * 11;
      // in_proj: (B*S,256) x (256,1024) -> silu(xi) f32, silu(res) f16
      wt_transpose_kernel<<<(H_ * 2 * DI_ + 255) / 256, 256, 0, stream>>>(
          P(head, bb + 0), inWt, H_, 2 * DI_);
      gemm_epi_kernel<<<dim3((unsigned)(MBS / 16), (2 * DI_) / 256), 512, 0, stream>>>(
          X16, inWt, P(head, bb + 1), H_, 0, XI, SRES, nullptr);
      // depthwise causal conv + silu
      {
        size_t n = MBS * DI_;
        conv_silu_kernel<<<(unsigned)((n + 255) / 256), 256, 0, stream>>>(
            XI, P(head, bb + 2), P(head, bb + 3), XI2);
      }
      // dt proj: (B*S,512) x (512,512) -> alpha f16 (aliases dead XI)
      wt_transpose_kernel<<<(DI_ * DI_ + 255) / 256, 256, 0, stream>>>(
          P(head, bb + 4), dtWt, DI_, DI_);
      gemm_epi_kernel<<<dim3((unsigned)(MBS / 16), DI_ / 256), 512, 0, stream>>>(
          XI2, dtWt, P(head, bb + 5), DI_, 1, nullptr, nullptr, ALPHA);
      // chunk-parallel scan, fused with (+xi*D) * silu(res)
      {
        int nt = B_ * CH_ * DI_;
        scanA_kernel<<<(nt + 255) / 256, 256, 0, stream>>>(XI2, ALPHA, Pb, Qb);
        scanB_kernel<<<(B_ * DI_ + 255) / 256, 256, 0, stream>>>(Pb, Qb, Yin);
        scanC_kernel<<<(nt + 255) / 256, 256, 0, stream>>>(
            XI2, ALPHA, SRES, P(head, bb + 6), Yin, Y16);
      }
      // out_proj + bias + residual + LN
      wt_transpose_kernel<<<(DI_ * H_ + 255) / 256, 256, 0, stream>>>(
          P(head, bb + 7), outWt, DI_, H_);
      gemm_ln_kernel<<<(unsigned)(MBS / 16), 512, 0, stream>>>(
          Y16, outWt, DI_, P(head, bb + 8), P(head, bb + 9), P(head, bb + 10),
          X, 0, X, X16);
    }

    head_mlp_kernel<<<1, 256, 0, stream>>>(
        X, ps, act,
        P(head, 48), P(head, 49), P(head, 50), P(head, 51),
        P(head, 52), P(head, 53), P(head, 54), P(head, 55),
        P(head, 56), P(head, 57), P(head, 58), P(head, 59),
        P(head, 60), P(head, 61),
        (float*)d_out + head * B_);
  }
}
